// Pct_21706764714409
// MI455X (gfx1250) — matrix operand order repaired and checked by reading, not compiled, not run on hardware
//
#include <hip/hip_runtime.h>
#include <hip/hip_bf16.h>

// Point Cloud Transformer forward for MI455X (gfx1250, wave32).
// Dense math via v_wmma_f32_16x16x32_f16; activations + weights stored f16;
// fragment loads vectorized to b128 on the full-tile fast path.

typedef _Float16 f16;
typedef __attribute__((ext_vector_type(16))) _Float16 v16h;
typedef __attribute__((ext_vector_type(8)))  _Float16 h8;
typedef __attribute__((ext_vector_type(8)))  float    v8f;
typedef __attribute__((ext_vector_type(4)))  float    f4;

union V16u { v16h v; h8 h[2]; };

__device__ __forceinline__ _Float16 to_h(float x)    { return (_Float16)x; }
__device__ __forceinline__ _Float16 to_h(_Float16 x) { return x; }

// 8 contiguous elements -> h8 (16B vector load; callers guarantee alignment)
__device__ __forceinline__ h8 load8h(const f16* p) { return *(const h8*)p; }
__device__ __forceinline__ h8 load8h(const float* p)
{
    const f4* q = (const f4*)p;
    f4 a = q[0], b = q[1];
    h8 r;
    #pragma unroll
    for (int i = 0; i < 4; ++i) { r[i] = (f16)a[i]; r[4+i] = (f16)b[i]; }
    return r;
}

// ---------------------------------------------------------------------------
// Generic batched WMMA GEMM:  Out[m,o] = sum_k A[m,k]*W[o,k] (+bias[o])
// AKC/WKC: operand is K-contiguous (enables b128 fragment loads).
// Fast path: full 16x16 tile, K%32==0 -> unconditional vector loads.
// Slow path (tiny GEMMs only): per-element guarded scalar loads.
// ---------------------------------------------------------------------------
template<typename TA, typename TW, typename TO, bool AKC, bool WKC>
__global__ void wmma_gemm(const TA* __restrict__ A, const TW* __restrict__ W,
                          TO* __restrict__ Out, const float* __restrict__ bias,
                          int M, int O, int K,
                          long long AsM, long long AsK,
                          long long WsO, long long WsK,
                          long long OsM, long long OsO,
                          long long Abb, long long Abh,
                          long long Wbb, long long Wbh,
                          long long Obb, long long Obh, int Hdiv)
{
    int z  = blockIdx.z;
    int bb = z / Hdiv, hh = z % Hdiv;
    const TA* Ab = A   + (long long)bb*Abb + (long long)hh*Abh;
    const TW* Wb = W   + (long long)bb*Wbb + (long long)hh*Wbh;
    TO*       Ob = Out + (long long)bb*Obb + (long long)hh*Obh;

    int wave = threadIdx.x >> 5;
    int lane = threadIdx.x & 31;
    int m0 = (blockIdx.y * (int)(blockDim.x >> 5) + wave) * 16;
    int n0 = blockIdx.x * 16;
    if (m0 >= M) return;                        // wave-uniform exit

    int lm = lane & 15, lh = lane >> 4;
    v8f acc = {0.f,0.f,0.f,0.f,0.f,0.f,0.f,0.f};

    bool full = (m0 + 16 <= M) && (n0 + 16 <= O) && ((K & 31) == 0);
    if (full) {
        const TA* arow = Ab + (long long)(m0 + lm) * AsM;
        const TW* wrow = Wb + (long long)(n0 + lm) * WsO;
        for (int k0 = 0; k0 < K; k0 += 32) {
            V16u af, bf;
            if constexpr (AKC) {
                af.h[0] = load8h(arow + k0 + 8*lh);
                af.h[1] = load8h(arow + k0 + 16 + 8*lh);
                __builtin_prefetch(arow + k0 + 64, 0, 0);
            } else {
                #pragma unroll
                for (int e = 0; e < 16; ++e) {
                    int v = e >> 1, p = e & 1;
                    int kk = (v < 4) ? (8*lh + 2*v + p) : (16 + 8*lh + 2*(v-4) + p);
                    af.v[e] = to_h(arow[(long long)(k0 + kk) * AsK]);
                }
            }
            if constexpr (WKC) {
                bf.h[0] = load8h(wrow + k0 + 8*lh);
                bf.h[1] = load8h(wrow + k0 + 16 + 8*lh);
            } else {
                #pragma unroll
                for (int e = 0; e < 16; ++e)
                    bf.v[e] = to_h(wrow[(long long)(k0 + 8*lh + e + (e >= 8 ? 8 : 0)) * WsK]);
            }
            acc = __builtin_amdgcn_wmma_f32_16x16x32_f16(false, af.v, false, bf.v,
                                                         (short)0, acc, false, false);
        }
        int n = n0 + lm;
        #pragma unroll
        for (int r = 0; r < 8; ++r) {
            int mi = m0 + r + 8*lh;
            float val = acc[r];
            if (bias) val += bias[n];
            Ob[(long long)mi*OsM + (long long)n*OsO] = (TO)val;
        }
    } else {
        int m = m0 + lm;  bool mok = (m < M);
        int n = n0 + lm;  bool nok = (n < O);
        for (int k0 = 0; k0 < K; k0 += 32) {
            V16u af, bf;
            #pragma unroll
            for (int e = 0; e < 16; ++e) {
                int v = e >> 1, p = e & 1;
                int kk = (v < 4) ? (8*lh + 2*v + p) : (16 + 8*lh + 2*(v-4) + p);
                int k = k0 + kk;
                af.v[e] = (mok && k < K)
                        ? to_h(Ab[(long long)m*AsM + (long long)k*AsK]) : (f16)0.f;
            }
            #pragma unroll
            for (int e = 0; e < 16; ++e) {
                int k = k0 + 8*lh + e + (e >= 8 ? 8 : 0);
                bf.v[e] = (nok && k < K)
                        ? to_h(Wb[(long long)n*WsO + (long long)k*WsK]) : (f16)0.f;
            }
            acc = __builtin_amdgcn_wmma_f32_16x16x32_f16(false, af.v, false, bf.v,
                                                         (short)0, acc, false, false);
        }
        #pragma unroll
        for (int r = 0; r < 8; ++r) {
            int mi = m0 + r + 8*lh;
            if (mi < M && nok) {
                float val = acc[r];
                if (bias) val += bias[n];
                Ob[(long long)mi*OsM + (long long)n*OsO] = (TO)val;
            }
        }
    }
}

// ---------------------------------------------------------------------------
// Fused neighbor-gather + (pts[idx]-cent | cent) concat + WMMA GEMM.
// Row m = (b*S + s)*kn + j.  D is a multiple of 32, so the concat boundary
// never splits a K-tile: phase 1 covers cols [0,D) (gathered minus center),
// phase 2 covers [D,2D) (broadcast center).  All loads are b128.
// M and O are multiples of 16 by construction.
// ---------------------------------------------------------------------------
__global__ void wmma_gemm_group(const f16* __restrict__ pts, const f16* __restrict__ cent,
                                const int* __restrict__ nidx, const f16* __restrict__ Wt,
                                f16* __restrict__ Out,
                                int S, int kn, int kmax, int Np, int D, int M, int O)
{
    int wave = threadIdx.x >> 5;
    int lane = threadIdx.x & 31;
    int m0 = (blockIdx.y * (int)(blockDim.x >> 5) + wave) * 16;
    int n0 = blockIdx.x * 16;
    if (m0 >= M) return;

    int lm = lane & 15, lh = lane >> 4;
    int K = 2 * D;
    int m = m0 + lm;
    int b   = m / (S * kn);
    int rem = m - b * (S * kn);
    int s   = rem / kn;
    int j   = rem - s * kn;
    int bs  = b * S + s;
    int gi  = nidx[(long long)bs * kmax + j];
    const f16* prow = pts  + ((long long)b * Np + gi) * D;
    const f16* crow = cent + (long long)bs * D;
    const f16* wrow = Wt   + (long long)(n0 + lm) * K;

    v8f acc = {0.f,0.f,0.f,0.f,0.f,0.f,0.f,0.f};

    for (int k0 = 0; k0 < D; k0 += 32) {          // cols [0,D): pts - cent
        V16u af, bf;
        h8 p0 = load8h(prow + k0 + 8*lh);
        h8 p1 = load8h(prow + k0 + 16 + 8*lh);
        h8 c0 = load8h(crow + k0 + 8*lh);
        h8 c1 = load8h(crow + k0 + 16 + 8*lh);
        af.h[0] = p0 - c0;
        af.h[1] = p1 - c1;
        bf.h[0] = load8h(wrow + k0 + 8*lh);
        bf.h[1] = load8h(wrow + k0 + 16 + 8*lh);
        acc = __builtin_amdgcn_wmma_f32_16x16x32_f16(false, af.v, false, bf.v,
                                                     (short)0, acc, false, false);
    }
    for (int k0 = 0; k0 < D; k0 += 32) {          // cols [D,2D): cent
        V16u af, bf;
        af.h[0] = load8h(crow + k0 + 8*lh);
        af.h[1] = load8h(crow + k0 + 16 + 8*lh);
        bf.h[0] = load8h(wrow + D + k0 + 8*lh);
        bf.h[1] = load8h(wrow + D + k0 + 16 + 8*lh);
        acc = __builtin_amdgcn_wmma_f32_16x16x32_f16(false, af.v, false, bf.v,
                                                     (short)0, acc, false, false);
    }
    int n = n0 + lm;
    #pragma unroll
    for (int r = 0; r < 8; ++r) {
        int mi = m0 + r + 8*lh;
        Out[(long long)mi*O + n] = (f16)acc[r];
    }
}

// ------------------------- weight f32 -> f16 pre-pass -----------------------
__global__ void cvt_w(const float* __restrict__ src, f16* __restrict__ dst, int n)
{
    int i = blockIdx.x * blockDim.x + threadIdx.x;
    if (i < n) dst[i] = (f16)src[i];
}

// ------------------------- training-mode BatchNorm -------------------------
template<typename T>
__global__ void bn_stats(const T* __restrict__ x, float* mean, float* rstd, int R, int C)
{
    int c = blockIdx.x, t = threadIdx.x;
    float s = 0.f, s2 = 0.f;
    for (int r = t; r < R; r += 256) {
        float v = (float)x[(long long)r * C + c];
        s += v; s2 += v * v;
    }
    __shared__ float sh[256], sh2[256];
    sh[t] = s; sh2[t] = s2; __syncthreads();
    for (int o = 128; o > 0; o >>= 1) {
        if (t < o) { sh[t] += sh[t+o]; sh2[t] += sh2[t+o]; }
        __syncthreads();
    }
    if (t == 0) {
        float m = sh[0] / (float)R;
        float v = sh2[0] / (float)R - m * m;
        mean[c] = m;
        rstd[c] = rsqrtf(v + 1e-5f);
    }
}

template<typename T>
__global__ void bn_act(T* __restrict__ x, const float* __restrict__ mean,
                       const float* __restrict__ rstd, long long total, int C, int act)
{
    long long i = (long long)blockIdx.x * blockDim.x + threadIdx.x;
    if (i >= total) return;
    int c = (int)(i % C);
    float v = ((float)x[i] - mean[c]) * rstd[c];
    if (act == 1)      v = v > 0.f ? v : 0.f;
    else if (act == 2) v = v > 0.f ? v : 0.2f * v;
    x[i] = (T)v;
}

// ------------------------- farthest point sampling -------------------------
__global__ void fps_kernel(const float* __restrict__ xyz, int* __restrict__ out, int N, int S)
{
    int b = blockIdx.x, t = threadIdx.x;
    __shared__ float dist[2048];
    __shared__ float sv[256];
    __shared__ int   si[256];
    __shared__ int   cur;
    const float* X = xyz + (long long)b * N * 3;
    for (int n = t; n < N; n += 256) dist[n] = 1e10f;
    if (t == 0) { out[b * S] = 0; cur = 0; }
    __syncthreads();
    for (int it = 1; it < S; ++it) {
        float cx = X[cur*3], cy = X[cur*3+1], cz = X[cur*3+2];
        float best = -1.f; int bi = N - 1;
        for (int n = t; n < N; n += 256) {
            float dx = X[n*3]-cx, dy = X[n*3+1]-cy, dz = X[n*3+2]-cz;
            float d = dx*dx + dy*dy + dz*dz;
            float dn = fminf(dist[n], d);
            dist[n] = dn;
            if (dn > best) { best = dn; bi = n; }
        }
        sv[t] = best; si[t] = bi; __syncthreads();
        for (int o = 128; o > 0; o >>= 1) {
            if (t < o) {
                if (sv[t+o] > sv[t] || (sv[t+o] == sv[t] && si[t+o] < si[t])) {
                    sv[t] = sv[t+o]; si[t] = si[t+o];
                }
            }
            __syncthreads();
        }
        if (t == 0) { out[b*S + it] = si[0]; cur = si[0]; }
        __syncthreads();
    }
}

// ------------------------- kNN (64 smallest, ascending) --------------------
__global__ void knn_kernel(const float* __restrict__ cent, const float* __restrict__ xyz,
                           int* __restrict__ idx, int S, int N, int K)
{
    int bs = blockIdx.x;
    int b  = bs / S;
    int t  = threadIdx.x;
    __shared__ float d[2048];
    __shared__ float sv[256];
    __shared__ int   si[256];
    const float* X = xyz + (long long)b * N * 3;
    float cx = cent[bs*3], cy = cent[bs*3+1], cz = cent[bs*3+2];
    for (int n = t; n < N; n += 256) {
        float dx = X[n*3]-cx, dy = X[n*3+1]-cy, dz = X[n*3+2]-cz;
        d[n] = dx*dx + dy*dy + dz*dz;
    }
    __syncthreads();
    for (int kk = 0; kk < K; ++kk) {
        float best = 1e30f; int bi = 0;
        for (int n = t; n < N; n += 256)
            if (d[n] < best) { best = d[n]; bi = n; }
        sv[t] = best; si[t] = bi; __syncthreads();
        for (int o = 128; o > 0; o >>= 1) {
            if (t < o) {
                if (sv[t+o] < sv[t] || (sv[t+o] == sv[t] && si[t+o] < si[t])) {
                    sv[t] = sv[t+o]; si[t] = si[t+o];
                }
            }
            __syncthreads();
        }
        if (t == 0) { idx[(long long)bs*K + kk] = si[0]; d[si[0]] = 1e30f; }
        __syncthreads();
    }
}

// ------------------------- small helpers ------------------------------------
__global__ void transpose_xyz(const float* __restrict__ x, float* __restrict__ xyz, int B, int N)
{
    long long i = (long long)blockIdx.x * blockDim.x + threadIdx.x;
    if (i >= (long long)B * N * 3) return;
    int d = (int)(i % 3);
    long long bn = i / 3;
    int n = (int)(bn % N);
    int b = (int)(bn / N);
    xyz[i] = x[((long long)b*3 + d)*N + n];
}

template<typename T>
__global__ void gather_rows(const T* __restrict__ src, const int* __restrict__ idx,
                            T* __restrict__ dst, int B, int N, int S, int D)
{
    long long i = (long long)blockIdx.x * blockDim.x + threadIdx.x;
    if (i >= (long long)B * S * D) return;
    int d = (int)(i % D);
    long long bs = i / D;
    int s = (int)(bs % S);
    int b = (int)(bs / S);
    dst[i] = src[((long long)b*N + idx[b*S + s])*D + d];
}

__global__ void maxpool_k(const f16* __restrict__ in, f16* __restrict__ out,
                          int BS, int kn, int C, int outLd, int colOff)
{
    long long i = (long long)blockIdx.x * blockDim.x + threadIdx.x;
    if (i >= (long long)BS * C) return;
    int c  = (int)(i % C);
    int bs = (int)(i / C);
    float m = -3.4e38f;
    for (int j = 0; j < kn; ++j)
        m = fmaxf(m, (float)in[((long long)bs*kn + j)*C + c]);
    out[(long long)bs*outLd + colOff + c] = (f16)m;
}

__global__ void ew_add(const f16* a, const f16* b, f16* o, long long n)
{
    long long i = (long long)blockIdx.x * blockDim.x + threadIdx.x;
    if (i < n) o[i] = (f16)((float)a[i] + (float)b[i]);
}
__global__ void ew_sub(const f16* a, const f16* b, f16* o, long long n)
{
    long long i = (long long)blockIdx.x * blockDim.x + threadIdx.x;
    if (i < n) o[i] = (f16)((float)a[i] - (float)b[i]);
}
__global__ void copy_cols(const f16* __restrict__ src, f16* __restrict__ dst,
                          long long R, int C, int ld, int colOff)
{
    long long i = (long long)blockIdx.x * blockDim.x + threadIdx.x;
    if (i >= R * C) return;
    long long r = i / C; int c = (int)(i % C);
    dst[r*ld + colOff + c] = src[i];
}

__global__ void softmax_rows(float* __restrict__ a, int L)
{
    long long row = blockIdx.x; int t = threadIdx.x;
    float* p = a + row * L;
    __shared__ float sh[256];
    float v = p[t];
    sh[t] = v; __syncthreads();
    for (int o = 128; o > 0; o >>= 1) { if (t < o) sh[t] = fmaxf(sh[t], sh[t+o]); __syncthreads(); }
    float mx = sh[0]; __syncthreads();
    float e = __expf(v - mx);
    sh[t] = e; __syncthreads();
    for (int o = 128; o > 0; o >>= 1) { if (t < o) sh[t] += sh[t+o]; __syncthreads(); }
    p[t] = e / sh[0];
}

__global__ void softmax_cols(float* __restrict__ a, int L)
{
    int blk = blockIdx.x;
    int z = blk / L, col = blk % L, t = threadIdx.x;
    float* p = a + (long long)z*L*L + col;
    __shared__ float sh[256];
    float v = p[(long long)t*L];
    sh[t] = v; __syncthreads();
    for (int o = 128; o > 0; o >>= 1) { if (t < o) sh[t] = fmaxf(sh[t], sh[t+o]); __syncthreads(); }
    float mx = sh[0]; __syncthreads();
    float e = __expf(v - mx);
    sh[t] = e; __syncthreads();
    for (int o = 128; o > 0; o >>= 1) { if (t < o) sh[t] += sh[t+o]; __syncthreads(); }
    p[(long long)t*L] = e / sh[0];
}

__global__ void rowmax(const f16* __restrict__ x, float* __restrict__ g, int B, int Nn, int C)
{
    long long i = (long long)blockIdx.x * blockDim.x + threadIdx.x;
    if (i >= (long long)B * C) return;
    int c = (int)(i % C);
    int b = (int)(i / C);
    float m = -3.4e38f;
    for (int n = 0; n < Nn; ++n)
        m = fmaxf(m, (float)x[((long long)b*Nn + n)*C + c]);
    g[i] = m;
}

// ------------------------- host-side launch helpers -------------------------
template<typename TA, typename TW, typename TO, bool AKC = true, bool WKC = true>
static void launch_gemm(hipStream_t st, const TA* A, const TW* W, TO* Out, const float* bias,
                        int M, int O, int K,
                        long long AsM, long long AsK, long long WsO, long long WsK,
                        long long OsM, long long OsO,
                        int Z = 1, int Hdiv = 1,
                        long long Abb = 0, long long Abh = 0,
                        long long Wbb = 0, long long Wbh = 0,
                        long long Obb = 0, long long Obh = 0)
{
    dim3 grid((O + 15) / 16, (M + 63) / 64, Z);
    wmma_gemm<TA, TW, TO, AKC, WKC><<<grid, 128, 0, st>>>(A, W, Out, bias, M, O, K,
        AsM, AsK, WsO, WsK, OsM, OsO, Abb, Abh, Wbb, Wbh, Obb, Obh, Hdiv);
}

template<typename TA, typename TW, typename TO>
static void launch_gemm_rm(hipStream_t st, const TA* A, const TW* W, TO* Out,
                           const float* bias, int M, int O, int K)
{
    launch_gemm<TA, TW, TO, true, true>(st, A, W, Out, bias, M, O, K,
                                        (long long)K, 1, (long long)K, 1, (long long)O, 1);
}

template<typename T>
static void launch_bn(hipStream_t st, T* x, float* mean, float* rstd, int R, int C, int act)
{
    bn_stats<T><<<C, 256, 0, st>>>(x, mean, rstd, R, C);
    long long tot = (long long)R * C;
    bn_act<T><<<(unsigned)((tot + 255) / 256), 256, 0, st>>>(x, mean, rstd, tot, C, act);
}

static inline unsigned gblk(long long n) { return (unsigned)((n + 255) / 256); }

// ---------------------------------------------------------------------------
extern "C" void kernel_launch(void* const* d_in, const int* in_sizes, int n_in,
                              void* d_out, int out_size, void* d_ws, size_t ws_size,
                              hipStream_t stream)
{
    (void)in_sizes; (void)n_in; (void)out_size; (void)ws_size;

    const float* x_in   = (const float*)d_in[0];
    const float* conv1w = (const float*)d_in[1];
    const float* conv2w = (const float*)d_in[2];
    const float* g0w1   = (const float*)d_in[3];
    const float* g0w2   = (const float*)d_in[4];
    const float* g0wc   = (const float*)d_in[5];
    const float* g1w1   = (const float*)d_in[6];
    const float* g1w2   = (const float*)d_in[7];
    const float* g1wc   = (const float*)d_in[8];
    const float* ptw    = (const float*)d_in[9];
    const float* posw   = (const float*)d_in[10];
    const float* posbias= (const float*)d_in[11];
    const float* sawqk  = (const float*)d_in[12];
    const float* sawv   = (const float*)d_in[13];
    const float* sabv   = (const float*)d_in[14];
    const float* sawt   = (const float*)d_in[15];
    const float* sabt   = (const float*)d_in[16];
    const float* fusew  = (const float*)d_in[17];
    const float* lin1w  = (const float*)d_in[18];
    const float* lin2w  = (const float*)d_in[19];
    const float* lin2b  = (const float*)d_in[20];
    const float* lin3w  = (const float*)d_in[21];
    const float* lin3b  = (const float*)d_in[22];

    const int B = 8, N1 = 2048, S1 = 512, S2 = 256, KMAX = 64;

    char* base = (char*)d_ws;
    size_t off = 0;
    auto alloc = [&](size_t bytes) -> void* {
        void* p = base + off;
        off = (off + bytes + 255) & ~(size_t)255;
        return p;
    };

    float* xyz     = (float*)alloc((size_t)B*N1*3*4);
    f16*   bufA    = (f16*)  alloc((size_t)64*1024*1024 + 4*1024*1024); // 68MB
    f16*   bufB    = (f16*)  alloc((size_t)64*1024*1024 + 4*1024*1024);
    f16*   points  = (f16*)  alloc((size_t)B*N1*64*2);
    int*   fidx    = (int*)  alloc((size_t)B*S1*4);
    float* newxyz  = (float*)alloc((size_t)B*S1*3*4);
    f16*   newpts  = (f16*)  alloc((size_t)B*S1*64*2);
    int*   idx0    = (int*)  alloc((size_t)B*S1*KMAX*4);
    f16*   pooled0 = (f16*)  alloc((size_t)B*S1*384*2);
    f16*   f0      = (f16*)  alloc((size_t)B*S1*128*2);
    int*   fidx2   = (int*)  alloc((size_t)B*S2*4);
    float* newxyz2 = (float*)alloc((size_t)B*S2*3*4);
    f16*   newpts2 = (f16*)  alloc((size_t)B*S2*128*2);
    int*   idx1    = (int*)  alloc((size_t)B*S2*KMAX*4);
    f16*   pooled1 = (f16*)  alloc((size_t)B*S2*768*2);
    f16*   f1      = (f16*)  alloc((size_t)B*S2*256*2);
    f16*   posb    = (f16*)  alloc((size_t)B*S2*256*2);
    f16*   hbuf    = (f16*)  alloc((size_t)B*S2*256*2);
    f16*   xabuf   = (f16*)  alloc((size_t)B*S2*256*2);
    f16*   Pbuf    = (f16*)  alloc((size_t)B*S2*256*2);
    f16*   Pvbuf   = (f16*)  alloc((size_t)B*S2*256*2);
    float* attn    = (float*)alloc((size_t)B*4*256*256*4);
    f16*   xrbuf   = (f16*)  alloc((size_t)B*S2*256*2);
    f16*   ubuf    = (f16*)  alloc((size_t)B*S2*256*2);
    f16*   x0buf   = (f16*)  alloc((size_t)B*S2*256*2);
    f16*   x1buf   = (f16*)  alloc((size_t)B*S2*256*2);
    f16*   catbuf  = (f16*)  alloc((size_t)B*S2*1280*2);
    f16*   fuseb   = (f16*)  alloc((size_t)B*S2*1024*2);
    float* gbuf    = (float*)alloc((size_t)B*1024*4);
    float* l1buf   = (float*)alloc((size_t)B*512*4);
    float* l2buf   = (float*)alloc((size_t)B*256*4);
    float* meanb   = (float*)alloc(2048*4);
    float* rstdb   = (float*)alloc(2048*4);

    // f16 weight copies (converted every call; deterministic)
    f16* wc1h  = (f16*)alloc((size_t)64*3*2);
    f16* wc2h  = (f16*)alloc((size_t)64*64*2);
    f16* wg01h = (f16*)alloc((size_t)128*128*2);
    f16* wg02h = (f16*)alloc((size_t)128*128*2);
    f16* wg0ch = (f16*)alloc((size_t)128*384*2);
    f16* wg11h = (f16*)alloc((size_t)256*256*2);
    f16* wg12h = (f16*)alloc((size_t)256*256*2);
    f16* wg1ch = (f16*)alloc((size_t)256*768*2);
    f16* wpth  = (f16*)alloc((size_t)256*256*2);
    f16* wposh = (f16*)alloc((size_t)256*3*2);
    f16* wqkh  = (f16*)alloc((size_t)4*256*256*2);
    f16* wvh   = (f16*)alloc((size_t)4*256*256*2);
    f16* wth   = (f16*)alloc((size_t)4*256*256*2);
    f16* wfuseh= (f16*)alloc((size_t)1024*1280*2);

    auto cvt = [&](const float* s, f16* d, int n) {
        cvt_w<<<gblk(n), 256, 0, stream>>>(s, d, n);
    };
    cvt(conv1w, wc1h, 64*3);       cvt(conv2w, wc2h, 64*64);
    cvt(g0w1, wg01h, 128*128);     cvt(g0w2, wg02h, 128*128);
    cvt(g0wc, wg0ch, 128*384);     cvt(g1w1, wg11h, 256*256);
    cvt(g1w2, wg12h, 256*256);     cvt(g1wc, wg1ch, 256*768);
    cvt(ptw, wpth, 256*256);       cvt(posw, wposh, 256*3);
    cvt(sawqk, wqkh, 4*256*256);   cvt(sawv, wvh, 4*256*256);
    cvt(sawt, wth, 4*256*256);     cvt(fusew, wfuseh, 1024*1280);

    // ---- input embedding: conv1 -> BN -> relu -> conv2 -> BN -> relu ----
    transpose_xyz<<<gblk((long long)B*N1*3), 256, 0, stream>>>(x_in, xyz, B, N1);
    f16* h64 = bufA;                               // [B*N1, 64] scratch
    launch_gemm<float,f16,f16>(stream, xyz, wc1h, h64, nullptr,
                               B*N1, 64, 3, 3, 1, 3, 1, 64, 1);
    launch_bn<f16>(stream, h64, meanb, rstdb, B*N1, 64, 1);
    launch_gemm_rm<f16,f16,f16>(stream, h64, wc2h, points, nullptr, B*N1, 64, 64);
    launch_bn<f16>(stream, points, meanb, rstdb, B*N1, 64, 1);

    // ---- stage 0: FPS(512) + kNN(64) + multiscale local op ----
    fps_kernel<<<B, 256, 0, stream>>>(xyz, fidx, N1, S1);
    gather_rows<float><<<gblk((long long)B*S1*3), 256, 0, stream>>>(xyz, fidx, newxyz, B, N1, S1, 3);
    gather_rows<f16><<<gblk((long long)B*S1*64), 256, 0, stream>>>(points, fidx, newpts, B, N1, S1, 64);
    knn_kernel<<<B*S1, 256, 0, stream>>>(newxyz, xyz, idx0, S1, N1, KMAX);

    for (int si = 0; si < 3; ++si) {
        int kn = 16 << si;
        int M  = B * S1 * kn;
        dim3 gg((128 + 15) / 16, (M + 63) / 64, 1);
        wmma_gemm_group<<<gg, 128, 0, stream>>>(points, newpts, idx0, wg01h, bufA,
                                                S1, kn, KMAX, N1, 64, M, 128);
        launch_bn<f16>(stream, bufA, meanb, rstdb, M, 128, 1);
        launch_gemm_rm<f16,f16,f16>(stream, bufA, wg02h, bufB, nullptr, M, 128, 128);
        launch_bn<f16>(stream, bufB, meanb, rstdb, M, 128, 1);
        maxpool_k<<<gblk((long long)B*S1*128), 256, 0, stream>>>(bufB, pooled0,
                                                                 B*S1, kn, 128, 384, si*128);
    }
    launch_gemm_rm<f16,f16,f16>(stream, pooled0, wg0ch, f0, nullptr, B*S1, 128, 384);
    launch_bn<f16>(stream, f0, meanb, rstdb, B*S1, 128, 1);

    // ---- stage 1: FPS(256) + kNN(64) + multiscale local op ----
    fps_kernel<<<B, 256, 0, stream>>>(newxyz, fidx2, S1, S2);
    gather_rows<float><<<gblk((long long)B*S2*3), 256, 0, stream>>>(newxyz, fidx2, newxyz2, B, S1, S2, 3);
    gather_rows<f16><<<gblk((long long)B*S2*128), 256, 0, stream>>>(f0, fidx2, newpts2, B, S1, S2, 128);
    knn_kernel<<<B*S2, 256, 0, stream>>>(newxyz2, newxyz, idx1, S2, S1, KMAX);

    for (int si = 0; si < 3; ++si) {
        int kn = 16 << si;
        int M  = B * S2 * kn;
        dim3 gg((256 + 15) / 16, (M + 63) / 64, 1);
        wmma_gemm_group<<<gg, 128, 0, stream>>>(f0, newpts2, idx1, wg11h, bufA,
                                                S2, kn, KMAX, S1, 128, M, 256);
        launch_bn<f16>(stream, bufA, meanb, rstdb, M, 256, 1);
        launch_gemm_rm<f16,f16,f16>(stream, bufA, wg12h, bufB, nullptr, M, 256, 256);
        launch_bn<f16>(stream, bufB, meanb, rstdb, M, 256, 1);
        maxpool_k<<<gblk((long long)B*S2*256), 256, 0, stream>>>(bufB, pooled1,
                                                                 B*S2, kn, 256, 768, si*256);
    }
    launch_gemm_rm<f16,f16,f16>(stream, pooled1, wg1ch, f1, nullptr, B*S2, 256, 768);
    launch_bn<f16>(stream, f1, meanb, rstdb, B*S2, 256, 1);

    // ---- positional embedding + feature projection ----
    launch_gemm<float,f16,f16>(stream, newxyz2, wposh, posb, posbias,
                               B*S2, 256, 3, 3, 1, 3, 1, 256, 1);
    launch_gemm_rm<f16,f16,f16>(stream, f1, wpth, hbuf, nullptr, B*S2, 256, 256);
    launch_bn<f16>(stream, hbuf, meanb, rstdb, B*S2, 256, 1);

    // ---- 4 offset self-attention layers (4 heads, head dim 64) ----
    const long long NC = (long long)S2 * 256;   // per-batch elements, channel-last
    f16* xprev = hbuf;
    for (int i = 0; i < 4; ++i) {
        const f16*   wqk = wqkh + (long long)i*256*256;
        const f16*   wv  = wvh  + (long long)i*256*256;
        const float* bv  = sabv + (long long)i*256;
        const f16*   wt  = wth  + (long long)i*256*256;
        const float* bt  = sabt + (long long)i*256;

        long long tot = (long long)B * NC;
        ew_add<<<gblk(tot), 256, 0, stream>>>(xprev, posb, xabuf, tot);

        launch_gemm_rm<f16,f16,f16>(stream, xabuf, wqk, Pbuf,  nullptr, B*S2, 256, 256);
        launch_gemm_rm<f16,f16,f16>(stream, xabuf, wv,  Pvbuf, bv,      B*S2, 256, 256);

        // attn[b,h,n2,n'] = q_h @ k_h : both views into Pbuf via strides
        launch_gemm<f16,f16,float,true,true>(stream, Pbuf, Pbuf, attn, nullptr,
                                   256, 256, 64,
                                   /*A q_h*/ 64, 1, /*W k_h^T*/ 256, 1,
                                   /*Out*/ 256, 1,
                                   /*Z*/ B*4, /*Hdiv*/ 4,
                                   NC, 64*256LL, NC, 64LL,
                                   4LL*65536, 65536LL);
        softmax_rows<<<B*4*256, 256, 0, stream>>>(attn, 256);
        softmax_cols<<<B*4*256, 256, 0, stream>>>(attn, 256);

        // xr[b,n,h*64+d] = sum_n' v_h[d,n'] * attn[b,h,n',n]
        launch_gemm<f16,float,f16,false,false>(stream, Pvbuf, attn, xrbuf, nullptr,
                                   64, 256, 256,
                                   /*A v_h*/ 1, 256, /*W a^T*/ 1, 256,
                                   /*Out*/ 1, 256,
                                   /*Z*/ B*4, /*Hdiv*/ 4,
                                   NC, 64LL, 4LL*65536, 65536LL,
                                   NC, 64LL);

        ew_sub<<<gblk(tot), 256, 0, stream>>>(xabuf, xrbuf, xrbuf, tot);
        launch_gemm_rm<f16,f16,f16>(stream, xrbuf, wt, ubuf, bt, B*S2, 256, 256);
        launch_bn<f16>(stream, ubuf, meanb, rstdb, B*S2, 256, 1);

        f16* xout = (i & 1) ? x1buf : x0buf;
        ew_add<<<gblk(tot), 256, 0, stream>>>(xabuf, ubuf, xout, tot);
        copy_cols<<<gblk(tot), 256, 0, stream>>>(xout, catbuf, (long long)B*S2, 256, 1280, i*256);
        xprev = xout;
    }
    copy_cols<<<gblk((long long)B*NC), 256, 0, stream>>>(f1, catbuf, (long long)B*S2, 256, 1280, 1024);

    // ---- fuse conv -> BN -> leaky -> global max ----
    launch_gemm_rm<f16,f16,f16>(stream, catbuf, wfuseh, fuseb, nullptr, B*S2, 1024, 1280);
    launch_bn<f16>(stream, fuseb, meanb, rstdb, B*S2, 1024, 2);
    rowmax<<<gblk((long long)B*1024), 256, 0, stream>>>(fuseb, gbuf, B, S2, 1024);

    // ---- classifier head (f32) ----
    launch_gemm_rm<float,float,float>(stream, gbuf, lin1w, l1buf, nullptr, B, 512, 1024);
    launch_bn<float>(stream, l1buf, meanb, rstdb, B, 512, 2);
    launch_gemm_rm<float,float,float>(stream, l1buf, lin2w, l2buf, lin2b, B, 256, 512);
    launch_bn<float>(stream, l2buf, meanb, rstdb, B, 256, 2);
    launch_gemm_rm<float,float,float>(stream, l2buf, lin3w, (float*)d_out, lin3b, B, 40, 256);
}
